// AttentionalPooler_43078521979142
// MI455X (gfx1250) — matrix operand order repaired and checked by reading, not compiled, not run on hardware
//
#include <hip/hip_runtime.h>

typedef __attribute__((ext_vector_type(16))) __bf16 bf16x16;
typedef __attribute__((ext_vector_type(8)))  __bf16 bf16x8;
typedef __attribute__((ext_vector_type(16))) unsigned short u16x16;
typedef __attribute__((ext_vector_type(8)))  float  floatx8;
typedef unsigned short u16;

// ---------- helpers ----------

__device__ __forceinline__ u16 f2bf(float f) {
    unsigned int u = __float_as_uint(f);
    u += 0x7FFFu + ((u >> 16) & 1u);          // round-to-nearest-even
    return (u16)(u >> 16);
}

__device__ __forceinline__ floatx8 wmma_bf16(bf16x16 a, bf16x16 b, floatx8 c) {
    return __builtin_amdgcn_wmma_f32_16x16x32_bf16(
        /*neg_a=*/false, a, /*neg_b=*/false, b,
        /*c_mod=*/(short)0, c, /*reuse_a=*/false, /*reuse_b=*/false);
}

// A fragment: 16x32 (MxK) bf16 from row-major [*, ld].
// Lane l: row m = l%16; holds K = (l/16)*16 + {0..15}.
__device__ __forceinline__ bf16x16 load_a_frag(const u16* base, int ld, int row0, int k0, int lane) {
    const int m  = row0 + (lane & 15);
    const int kh = (lane >> 4) * 16;
    const bf16x8 lo = *(const bf16x8*)(base + (size_t)m * ld + k0 + kh);
    const bf16x8 hi = *(const bf16x8*)(base + (size_t)m * ld + k0 + 8 + kh);
    return __builtin_shufflevector(lo, hi, 0, 1, 2, 3, 4, 5, 6, 7, 8, 9, 10, 11, 12, 13, 14, 15);
}

// B fragment: 32x16 (KxN) bf16 from B^T row-major [N, ld=K].
// Lane l: col n = l%16; holds K = (l/16)*16 + {0..15}, contiguous in B^T row.
__device__ __forceinline__ bf16x16 load_b_frag(const u16* base, int ld, int col0, int k0, int lane) {
    const int n  = col0 + (lane & 15);
    const int kk = k0 + (lane >> 4) * 16;
    return *(const bf16x16*)(base + (size_t)n * ld + kk);
}

// max-reduction across one 16-lane half (xor masks < 16 never cross halves)
__device__ __forceinline__ float rmax16(float v) {
    v = fmaxf(v, __shfl_xor(v, 1, 32));
    v = fmaxf(v, __shfl_xor(v, 2, 32));
    v = fmaxf(v, __shfl_xor(v, 4, 32));
    v = fmaxf(v, __shfl_xor(v, 8, 32));
    return v;
}

// ---------- kernel 1: weight transpose + fp32 -> bf16 ----------
// W [K,N] row-major  ->  Wt [N,K] row-major (bf16)
__global__ void wtrans_kernel(const float* __restrict__ W, u16* __restrict__ Wt, int K, int N) {
    int idx = blockIdx.x * blockDim.x + threadIdx.x;
    if (idx >= K * N) return;
    int k = idx / N, n = idx % N;
    Wt[(size_t)n * K + k] = f2bf(W[idx]);
}

// ---------- kernel 2: wave-per-row LayerNorm (width 1024) -> bf16 ----------
__global__ __launch_bounds__(256) void ln_kernel(const float* __restrict__ in,
                                                 const float* __restrict__ g,
                                                 const float* __restrict__ bta,
                                                 u16* __restrict__ outp, int rows) {
    const int gw   = (blockIdx.x * 256 + threadIdx.x) >> 5;
    const int lane = threadIdx.x & 31;
    if (gw >= rows) return;
    const float4* rp = (const float4*)(in + (size_t)gw * 1024);
    float4 v[8];
    float s = 0.f, s2 = 0.f;
#pragma unroll
    for (int i = 0; i < 8; ++i) {
        float4 t = rp[lane + i * 32];
        v[i] = t;
        s  += t.x + t.y + t.z + t.w;
        s2 += t.x * t.x + t.y * t.y + t.z * t.z + t.w * t.w;
    }
#pragma unroll
    for (int m = 1; m < 32; m <<= 1) { s += __shfl_xor(s, m, 32); s2 += __shfl_xor(s2, m, 32); }
    const float mu  = s * (1.0f / 1024.0f);
    const float var = s2 * (1.0f / 1024.0f) - mu * mu;
    const float inv = rsqrtf(var + 1e-5f);
    ushort4* op = (ushort4*)(outp + (size_t)gw * 1024);
#pragma unroll
    for (int i = 0; i < 8; ++i) {
        const int c4 = (lane + i * 32) * 4;
        float4 t = v[i];
        ushort4 o;
        o.x = f2bf((t.x - mu) * inv * g[c4 + 0] + bta[c4 + 0]);
        o.y = f2bf((t.y - mu) * inv * g[c4 + 1] + bta[c4 + 1]);
        o.z = f2bf((t.z - mu) * inv * g[c4 + 2] + bta[c4 + 2]);
        o.w = f2bf((t.w - mu) * inv * g[c4 + 3] + bta[c4 + 3]);
        op[lane + i * 32] = o;
    }
}

// ---------- kernel 3: bf16 WMMA GEMM, wave computes (MT*16)x64 tile ----------
// A [M,K] row-major bf16, B = "B^T" [N,K] row-major bf16 (weights pre-transposed,
// or an activation matrix used as the transposed operand).
// blockIdx.z batches the B operand / output (stride in elements).
// BF16OUT=true : obf[m*N+n] = bf16(acc*scale) ; else ofp[m*N+n] = acc*scale (fp32)
template <bool BF16OUT, int MT>
__global__ __launch_bounds__(256) void gemm_kernel(const u16* __restrict__ A,
                                                   const u16* __restrict__ Bmat,
                                                   int N, int K,
                                                   u16* __restrict__ obf,
                                                   float* __restrict__ ofp,
                                                   float scale,
                                                   size_t batchB, size_t batchO) {
    const int lane = threadIdx.x & 31;
    const int wave = threadIdx.x >> 5;
    const int row0 = blockIdx.y * (MT * 128) + wave * (MT * 16);
    const int col0 = blockIdx.x * 64;
    const u16* B = Bmat + (size_t)blockIdx.z * batchB;

    floatx8 c[MT * 4];
#pragma unroll
    for (int i = 0; i < MT * 4; ++i) c[i] = {};

    for (int k = 0; k < K; k += 32) {
        bf16x16 a[MT];
#pragma unroll
        for (int mi = 0; mi < MT; ++mi)
            a[mi] = load_a_frag(A, K, row0 + mi * 16, k, lane);
        if (k + 32 < K)
            __builtin_prefetch(A + (size_t)(row0 + (lane & 15)) * K + k + 32);
#pragma unroll
        for (int nb = 0; nb < 4; ++nb) {
            bf16x16 b = load_b_frag(B, K, col0 + nb * 16, k, lane);
#pragma unroll
            for (int mi = 0; mi < MT; ++mi)
                c[mi * 4 + nb] = wmma_bf16(a[mi], b, c[mi * 4 + nb]);
        }
    }

    const int rl = (lane >> 4) * 8;
    const int nl = lane & 15;
#pragma unroll
    for (int mi = 0; mi < MT; ++mi)
#pragma unroll
        for (int nb = 0; nb < 4; ++nb)
#pragma unroll
            for (int j = 0; j < 8; ++j) {
                const int m = row0 + mi * 16 + rl + j;
                const int n = col0 + nb * 16 + nl;
                const float v = c[mi * 4 + nb][j] * scale;
                if (BF16OUT)
                    obf[(size_t)blockIdx.z * batchO + (size_t)m * N + n] = f2bf(v);
                else
                    ofp[(size_t)blockIdx.z * batchO + (size_t)m * N + n] = v;
            }
}

// ---------- kernel 4: flash attention per (b,h), 16 queries per wave ----------
// qproj [256,1024] bf16 (pre-scaled), kbuf [b][4096][1024] bf16,
// vt [b][1024][4096] bf16 (d-major), aout [b*256][1024] bf16.
//
// Online softmax with two cheapening tricks:
//  * block-max: each lane-half uses the max over its 8 rows x 32 keys as the
//    shared running max m (valid upper bound, consistent per row, monotone) —
//    reduces the cross-lane reduction to ONE 4-stage shuffle per tile and makes
//    the rescale factor a scalar.
//  * row-sums of exp(P) accumulated by an extra WMMA against an all-ones B
//    fragment (every output column equals the row-sum) — no sum shuffles.
__global__ __launch_bounds__(256) void attn_kernel(const u16* __restrict__ qproj,
                                                   const u16* __restrict__ kbuf,
                                                   const u16* __restrict__ vt,
                                                   u16* __restrict__ aout) {
    const int lane = threadIdx.x & 31;
    const int wave = threadIdx.x >> 5;
    const int bh = blockIdx.x >> 1;
    const int b = bh >> 4, h = bh & 15;
    const int q0 = (blockIdx.x & 1) * 128 + wave * 16;

    __shared__ alignas(16) u16 plds[8][16 * 32];   // per-wave 16x32 bf16 P staging
    u16* myp = &plds[wave][0];

    const u16* qb = qproj + h * 64;
    const u16* kb = kbuf + (size_t)b * 4096 * 1024 + h * 64;
    const u16* vb = vt + ((size_t)b * 1024 + h * 64) * 4096;

    const bf16x16 qa0 = load_a_frag(qb, 1024, q0, 0, lane);
    const bf16x16 qa1 = load_a_frag(qb, 1024, q0, 32, lane);

    // all-ones bf16 B fragment (bf16 1.0 == 0x3F80)
    u16x16 ob;
#pragma unroll
    for (int i = 0; i < 16; ++i) ob[i] = 0x3F80u;
    const bf16x16 onesb = __builtin_bit_cast(bf16x16, ob);

    floatx8 acc[4];
#pragma unroll
    for (int nb = 0; nb < 4; ++nb) acc[nb] = {};
    floatx8 accl = {};                 // row-sum accumulator (P @ ones)
    float mold = -1e30f;               // running block max (per lane-half)

    for (int kt = 0; kt < 4096; kt += 32) {
        floatx8 c0 = {}, c1 = {};
        c0 = wmma_bf16(qa0, load_b_frag(kb, 1024, kt,      0,  lane), c0);
        c0 = wmma_bf16(qa1, load_b_frag(kb, 1024, kt,      32, lane), c0);
        c1 = wmma_bf16(qa0, load_b_frag(kb, 1024, kt + 16, 0,  lane), c1);
        c1 = wmma_bf16(qa1, load_b_frag(kb, 1024, kt + 16, 32, lane), c1);

        // per-lane max over all 16 sim values, then one cross-lane reduction
        float lm = fmaxf(c0[0], c1[0]);
#pragma unroll
        for (int j = 1; j < 8; ++j) lm = fmaxf(lm, fmaxf(c0[j], c1[j]));
        const float mnew  = fmaxf(mold, rmax16(lm));
        const float alpha = __expf(mold - mnew);
        mold = mnew;

#pragma unroll
        for (int j = 0; j < 8; ++j) {
            const float p0 = __expf(c0[j] - mnew);
            const float p1 = __expf(c1[j] - mnew);
            const int row = (lane >> 4) * 8 + j;
            myp[row * 32 +      (lane & 15)] = f2bf(p0);
            myp[row * 32 + 16 + (lane & 15)] = f2bf(p1);
        }
#pragma unroll
        for (int j = 0; j < 8; ++j) {
            acc[0][j] = acc[0][j] * alpha;
            acc[1][j] = acc[1][j] * alpha;
            acc[2][j] = acc[2][j] * alpha;
            acc[3][j] = acc[3][j] * alpha;
            accl[j]   = accl[j]   * alpha;
        }

        // re-read P as an A-fragment (wave-private LDS; in-order DS pipe)
        bf16x16 pa;
        {
            const int m  = lane & 15;
            const int kh = (lane >> 4) * 16;
            const bf16x8 lo = *(const bf16x8*)(myp + m * 32 + kh);
            const bf16x8 hi = *(const bf16x8*)(myp + m * 32 + 8 + kh);
            pa = __builtin_shufflevector(lo, hi, 0, 1, 2, 3, 4, 5, 6, 7,
                                         8, 9, 10, 11, 12, 13, 14, 15);
        }

#pragma unroll
        for (int nb = 0; nb < 4; ++nb)
            acc[nb] = wmma_bf16(pa, load_b_frag(vb, 4096, nb * 16, kt, lane), acc[nb]);
        accl = wmma_bf16(pa, onesb, accl);
    }

#pragma unroll
    for (int j = 0; j < 8; ++j) {
        const float inv  = 1.0f / accl[j];
        const int   qrow = q0 + (lane >> 4) * 8 + j;
        u16* orow = aout + ((size_t)b * 256 + qrow) * 1024 + h * 64 + (lane & 15);
#pragma unroll
        for (int nb = 0; nb < 4; ++nb)
            orow[nb * 16] = f2bf(acc[nb][j] * inv);
    }
}

// ---------- launch ----------
extern "C" void kernel_launch(void* const* d_in, const int* in_sizes, int n_in,
                              void* d_out, int out_size, void* d_ws, size_t ws_size,
                              hipStream_t stream) {
    (void)in_sizes; (void)n_in; (void)out_size; (void)ws_size;

    const float* x      = (const float*)d_in[0];   // [8,4096,1024]
    const float* query  = (const float*)d_in[1];   // [256,1024]
    const float* Wq     = (const float*)d_in[2];   // [1024,1024] (in,out)
    const float* Wkv    = (const float*)d_in[3];   // [1024,2048]
    const float* Wout   = (const float*)d_in[4];   // [1024,1024]
    const float* ln_q_g = (const float*)d_in[5];
    const float* ln_q_b = (const float*)d_in[6];
    const float* ln_k_g = (const float*)d_in[7];
    const float* ln_k_b = (const float*)d_in[8];
    float* out = (float*)d_out;                    // [8,256,1024]

    char* w = (char*)d_ws;
    size_t off = 0;
    auto alloc = [&](size_t bytes) -> char* {
        char* p = w + off;
        off += (bytes + 255) & ~(size_t)255;
        return p;
    };
    u16* xk    = (u16*)alloc((size_t)32768 * 1024 * 2);       // LN(x) bf16
    u16* qln   = (u16*)alloc((size_t)256 * 1024 * 2);         // LN(query) bf16
    u16* wqT   = (u16*)alloc((size_t)1024 * 1024 * 2);        // Wq^T bf16
    u16* wkvT  = (u16*)alloc((size_t)2048 * 1024 * 2);        // Wkv^T bf16 (K rows 0..1023, V rows 1024..2047)
    u16* woutT = (u16*)alloc((size_t)1024 * 1024 * 2);        // Wout^T bf16
    u16* qproj = (u16*)alloc((size_t)256 * 1024 * 2);         // scaled Q bf16
    u16* kbuf  = (u16*)alloc((size_t)8 * 4096 * 1024 * 2);    // K bf16 [b*4096][1024]
    u16* vtb   = (u16*)alloc((size_t)8 * 1024 * 4096 * 2);    // V^T bf16 [b][1024][4096]
    u16* aout  = (u16*)alloc((size_t)2048 * 1024 * 2);        // attention out bf16

    // weights -> bf16 transposed
    wtrans_kernel<<<(1024 * 1024 + 255) / 256, 256, 0, stream>>>(Wq,   wqT,   1024, 1024);
    wtrans_kernel<<<(1024 * 2048 + 255) / 256, 256, 0, stream>>>(Wkv,  wkvT,  1024, 2048);
    wtrans_kernel<<<(1024 * 1024 + 255) / 256, 256, 0, stream>>>(Wout, woutT, 1024, 1024);

    // layernorms
    ln_kernel<<<32768 / 8, 256, 0, stream>>>(x,     ln_k_g, ln_k_b, xk,  32768);
    ln_kernel<<<256 / 8,   256, 0, stream>>>(query, ln_q_g, ln_q_b, qln, 256);

    // q = LN(q) @ Wq, scaled by dim_head^-0.5 = 0.125   (32-row tiles, M=256)
    gemm_kernel<true, 2><<<dim3(16, 1, 1), 256, 0, stream>>>(
        qln, wqT, 1024, 1024, qproj, nullptr, 0.125f, 0, 0);

    // K = LN(x) @ Wk  -> row-major [32768,1024]
    gemm_kernel<true, 4><<<dim3(16, 64, 1), 256, 0, stream>>>(
        xk, wkvT, 1024, 1024, kbuf, nullptr, 1.0f, 0, 0);

    // V^T[b] = Wv^T (A, M=1024 d-rows) x LN(x)[b] (as B^T, N=4096 keys)
    // -> row-major [1024,4096] per batch: exactly the d-major layout attention wants.
    gemm_kernel<true, 4><<<dim3(64, 2, 8), 256, 0, stream>>>(
        wkvT + (size_t)1024 * 1024, xk, 4096, 1024, vtb, nullptr, 1.0f,
        /*batchB=*/(size_t)4096 * 1024, /*batchO=*/(size_t)1024 * 4096);

    // attention
    attn_kernel<<<256, 256, 0, stream>>>(qproj, kbuf, vtb, aout);

    // out = attn_out @ Wout  (fp32 result)
    gemm_kernel<false, 4><<<dim3(16, 4, 1), 256, 0, stream>>>(
        aout, woutT, 1024, 1024, nullptr, out, 1.0f, 0, 0);
}
